// MultiHeadAttention_10780367913667
// MI455X (gfx1250) — matrix operand order repaired and checked by reading, not compiled, not run on hardware
//
#include <hip/hip_runtime.h>

// ---------------------------------------------------------------------------
// Multi-head attention for MI455X (gfx1250), bf16 WMMA pipeline.
//   q = Q@Wq+bq ; k = K@Wk+bk ; v = V@Wv+bv   (bf16 WMMA, fp32 accum)
//   flash attention per (b,h,16-query tile)   (bf16 WMMA, fp32 softmax,
//                                              TDM tensor_load_to_lds staging)
//   out = ctx@Wo+bo                           (bf16 WMMA, fp32 out)
// ---------------------------------------------------------------------------

#define D_MODEL   1024
#define NUM_HEADS 16
#define D_K       64
#define SEQ       2048
#define BATCH     2
#define MTOK      (BATCH * SEQ)          // 4096 token rows
#define SM_SCALE  0.125f                 // 1/sqrt(64)

typedef unsigned short     u16;
typedef unsigned int       u32;
typedef unsigned long long u64;

typedef __attribute__((ext_vector_type(16))) __bf16 v16bf;
typedef __attribute__((ext_vector_type(8)))  float  v8f;
typedef __attribute__((ext_vector_type(4)))  unsigned int v4u;
typedef __attribute__((ext_vector_type(8)))  int    v8i;
typedef __attribute__((ext_vector_type(4)))  int    v4i;

union Frag {            // one WMMA 16-bit operand: 16 bf16 in 8 VGPRs
  v16bf bf;
  u32   u[8];
};

__device__ __forceinline__ u16 f2bf(float f) {
  u32 u = __float_as_uint(f);
  return (u16)((u + 0x7FFFu + ((u >> 16) & 1u)) >> 16);   // RNE
}

// load 8 contiguous bf16 (16 bytes, 16B-aligned) into fragment half `slot`
__device__ __forceinline__ void load8(Frag& f, int slot, const u16* p) {
  uint4 d = *(const uint4*)p;
  f.u[slot * 4 + 0] = d.x;
  f.u[slot * 4 + 1] = d.y;
  f.u[slot * 4 + 2] = d.z;
  f.u[slot * 4 + 3] = d.w;
}

__device__ __forceinline__ v8f wmma_bf16(const Frag& a, const Frag& b, v8f c) {
  return __builtin_amdgcn_wmma_f32_16x16x32_bf16(
      /*neg_a=*/false, a.bf, /*neg_b=*/false, b.bf,
      /*c_mod=*/(short)0, c, /*reuse_a=*/false, /*reuse_b=*/false);
}

// ---------------------------------------------------------------------------
// TDM: issue tensor_load_to_lds of a 2-D bf16 tile (tile1 rows x tile0 elems,
// row stride stride0 elems) from global gaddr into LDS offset lds_off.
// D# packing per CDNA5 ISA:
//   group0: [1:0]=count=1, [63:32]=lds_addr, [120:64]=global_addr, [127:126]=type=2
//   group1: [17:16]=data_size=1 (2B), [79:48]=tensor_dim0, [111:80]=tensor_dim1,
//           [127:112]=tile_dim0, [143:128]=tile_dim1, [207:160]=tensor_dim0_stride
__device__ __forceinline__ void tdm_load_tile(u32 lds_off, u64 gaddr,
                                              u32 tile0, u32 tile1,
                                              u32 td0, u32 td1, u64 stride0) {
  v4u g0;
  g0[0] = 1u;                                          // count=1, user descriptor
  g0[1] = lds_off;                                     // lds_addr (bytes)
  g0[2] = (u32)gaddr;                                  // global_addr[31:0]
  g0[3] = ((u32)(gaddr >> 32) & 0x01FFFFFFu) | (2u << 30);  // addr[56:32] | type=2
  v8i g1;
  g1[0] = (int)(1u << 16);                             // workgroup_mask=0, data_size=1 (2B)
  g1[1] = (int)((td0 & 0xFFFFu) << 16);                // tensor_dim0[15:0] at bit 48
  g1[2] = (int)(((td0 >> 16) & 0xFFFFu) | ((td1 & 0xFFFFu) << 16));
  g1[3] = (int)(((td1 >> 16) & 0xFFFFu) | ((tile0 & 0xFFFFu) << 16));
  g1[4] = (int)(tile1 & 0xFFFFu);                      // tile_dim1, tile_dim2=0
  g1[5] = (int)(u32)(stride0 & 0xFFFFFFFFu);           // tensor_dim0_stride[31:0]
  g1[6] = (int)(u32)((stride0 >> 32) & 0xFFFFu);       // stride[47:32], dim1_stride=0
  g1[7] = 0;
  v4i g2 = {};                                         // dims 2/3 unused (zero)
  v4i g3 = {};
#if defined(__clang_major__) && (__clang_major__ >= 23)
  v8i g4 = {};
  __builtin_amdgcn_tensor_load_to_lds(g0, g1, g2, g3, g4, 0);
#else
  __builtin_amdgcn_tensor_load_to_lds(g0, g1, g2, g3, 0);
#endif
}

// ---------------------------------------------------------------------------
// elementwise f32 -> bf16
__global__ void mha_cvt_bf16(const float* __restrict__ in, u16* __restrict__ out, int n) {
  int i = blockIdx.x * blockDim.x + threadIdx.x;
  if (i < n) out[i] = f2bf(in[i]);
}

// W[k*N+n] (f32, [K,N]) -> Wt[n*K+k] (bf16, [N,K]) so B-operand loads are contiguous in k
__global__ void mha_cvtT_bf16(const float* __restrict__ W, u16* __restrict__ Wt, int K, int N) {
  int i = blockIdx.x * blockDim.x + threadIdx.x;
  if (i >= K * N) return;
  int k = i / N, n = i - k * N;
  Wt[(size_t)n * K + k] = f2bf(W[i]);
}

// ---------------------------------------------------------------------------
// GEMM: C[M,N] = A[M,K](bf16,row) @ Wt[N,K](bf16,row)^T + bias
// one wave -> 16x64 tile (A fragment reused for 4 N-subtiles).
// LAYOUT 0: bf16 out, q/k head-split [B,H,S,64]
// LAYOUT 1: bf16 out, v transposed  [B,H,64,S]
// LAYOUT 2: f32 out, plain [M,N] (final projection -> d_out)
template <int LAYOUT, typename OUT_T>
__global__ __launch_bounds__(128) void mha_gemm_bf16(
    const u16* __restrict__ A, const u16* __restrict__ Bt,
    const float* __restrict__ bias, OUT_T* __restrict__ out,
    int M, int N, int K) {
  const int wave = (blockIdx.x * blockDim.x + threadIdx.x) >> 5;
  const int lane = threadIdx.x & 31;
  const int tilesN = N >> 6;                 // 64-wide wave tiles
  const int m0 = (wave / tilesN) << 4;
  const int n0 = (wave % tilesN) << 6;
  if (m0 >= M) return;
  const int mrow = lane & 15;
  const int hi   = lane >> 4;

  v8f acc[4] = {{}, {}, {}, {}};

  const u16* arow = A + (size_t)(m0 + mrow) * K;
  for (int k0 = 0; k0 < K; k0 += 32) {
    Frag a;
    const u16* ap = arow + k0 + hi * 8;      // A 16x32: runs [k0+hi*8, +8) and [k0+16+hi*8, +8)
    load8(a, 0, ap);
    load8(a, 1, ap + 16);
    __builtin_prefetch(ap + 32, 0, 1);       // global_prefetch_b8, next K step
#pragma unroll
    for (int t = 0; t < 4; ++t) {
      Frag b;                                // B 32x16: lane=n, runs [k0+hi*8, +8) and [k0+16+hi*8, +8)
      const u16* bp = Bt + (size_t)(n0 + t * 16 + mrow) * K + k0 + hi * 8;
      load8(b, 0, bp);
      load8(b, 1, bp + 16);
      acc[t] = wmma_bf16(a, b, acc[t]);
    }
  }

#pragma unroll
  for (int t = 0; t < 4; ++t) {
    const int col = n0 + t * 16 + mrow;
    const float bv = bias[col];
#pragma unroll
    for (int r = 0; r < 8; ++r) {
      const int row = m0 + r + hi * 8;       // C/D: row = r + 8*hi, col = lane&15
      const float val = acc[t][r] + bv;
      if (LAYOUT == 2) {
        ((float*)out)[(size_t)row * N + col] = val;
      } else {
        const int b = row >> 11, s = row & 2047;
        const int h = col >> 6,  d = col & 63;
        size_t idx;
        if (LAYOUT == 0) idx = (((size_t)(b * NUM_HEADS + h) * SEQ) + s) * D_K + d;
        else             idx = (((size_t)(b * NUM_HEADS + h) * D_K) + d) * SEQ + s;
        ((u16*)out)[idx] = f2bf(val);
      }
    }
  }
}

// ---------------------------------------------------------------------------
// Flash attention: one wave per (b, h, 16-query tile). Key blocks of 32.
// K/V tiles staged into LDS by the Tensor Data Mover, double-buffered on
// TENSORcnt; fragments then read from LDS (ds_load_b128).
// qp,kp: [B,H,S,64] bf16 ; vT: [B,H,64,S] bf16 ; ctx: [MTOK, D_MODEL] bf16
__global__ __launch_bounds__(64) void mha_flash(
    const u16* __restrict__ qp, const u16* __restrict__ kp,
    const u16* __restrict__ vT, u16* __restrict__ ctx) {
  // per wave: K tile [32 keys][64 d], V tile [64 d][32 keys], double buffered
  __shared__ __align__(16) u16  sK[2][2][32 * 64];
  __shared__ __align__(16) u16  sV[2][2][64 * 32];
  __shared__ float sP[2][16 * 33];           // exp(S) staging, stride 33
  const int wib  = threadIdx.x >> 5;         // wave in block (2 waves)
  const int lane = threadIdx.x & 31;
  const int gw   = blockIdx.x * 2 + wib;     // 0 .. B*H*128-1
  const int qt   = gw & 127;
  const int bh   = gw >> 7;                  // b*16+h
  const int mrow = lane & 15;
  const int hi   = lane >> 4;

  const u16* q = qp + (size_t)bh * SEQ * D_K;
  const u16* k = kp + (size_t)bh * SEQ * D_K;
  const u16* v = vT + (size_t)bh * D_K * SEQ;
  float* myLds = sP[wib];

  // Q A-fragments for d-halves [0,32) and [32,64) (loaded once, from global)
  Frag aq[2];
#pragma unroll
  for (int st = 0; st < 2; ++st) {
    const u16* ap = q + (size_t)(qt * 16 + mrow) * D_K + st * 32 + hi * 8;
    load8(aq[st], 0, ap);
    load8(aq[st], 1, ap + 16);
  }

  v8f o[4] = {{}, {}, {}, {}};
  float mrun[8], lrun[8];
#pragma unroll
  for (int r = 0; r < 8; ++r) { mrun[r] = -3.0e38f; lrun[r] = 0.0f; }

  // TDM issue of key-block kbi into LDS buffer buf (2 tensor ops per tile)
  auto issue_tile = [&](int kbi, int buf) {
    // K tile: 32 contiguous rows of 64 bf16 -> flat 2048-element copy
    tdm_load_tile((u32)(size_t)&sK[wib][buf][0],
                  (u64)(size_t)(k + (size_t)kbi * 32 * D_K),
                  /*tile0=*/2048, /*tile1=*/1, /*td0=*/2048, /*td1=*/1,
                  /*stride0=*/2048);
    // V tile: 64 rows (d) x 32 keys, row stride SEQ in vT
    tdm_load_tile((u32)(size_t)&sV[wib][buf][0],
                  (u64)(size_t)(v + (size_t)kbi * 32),
                  /*tile0=*/32, /*tile1=*/64, /*td0=*/SEQ, /*td1=*/D_K,
                  /*stride0=*/SEQ);
  };

  issue_tile(0, 0);

  for (int kb = 0; kb < 64; ++kb) {
    const int buf = kb & 1;
    if (kb < 63) {
      issue_tile(kb + 1, buf ^ 1);
      __builtin_amdgcn_s_wait_tensorcnt(2);  // retire tile kb (TDM in-order per wave)
    } else {
      __builtin_amdgcn_s_wait_tensorcnt(0);
    }
    asm volatile("" ::: "memory");           // don't hoist LDS reads above the wait
    const u16* kT = &sK[wib][buf][0];
    const u16* vTl = &sV[wib][buf][0];

    // scores: two 16-key subtiles, K(d)=64 via 2 WMMA steps each
    v8f s0 = {}, s1 = {};
#pragma unroll
    for (int st = 0; st < 2; ++st) {
      Frag bk0, bk1;
      const u16* bp0 = kT + (mrow)      * D_K + st * 32 + hi * 8;
      load8(bk0, 0, bp0);  load8(bk0, 1, bp0 + 16);
      const u16* bp1 = kT + (16 + mrow) * D_K + st * 32 + hi * 8;
      load8(bk1, 0, bp1);  load8(bk1, 1, bp1 + 16);
      s0 = wmma_bf16(aq[st], bk0, s0);
      s1 = wmma_bf16(aq[st], bk1, s1);
    }

    // online softmax update (rows r+8*hi live across the 16 lanes of this half)
    float e0[8], e1[8];
#pragma unroll
    for (int r = 0; r < 8; ++r) {
      float a0 = s0[r] * SM_SCALE;
      float a1 = s1[r] * SM_SCALE;
      float cand = fmaxf(a0, a1);
      cand = fmaxf(cand, __shfl_xor(cand, 1, 32));
      cand = fmaxf(cand, __shfl_xor(cand, 2, 32));
      cand = fmaxf(cand, __shfl_xor(cand, 4, 32));
      cand = fmaxf(cand, __shfl_xor(cand, 8, 32));
      const float mnew  = fmaxf(mrun[r], cand);
      const float alpha = __expf(mrun[r] - mnew);
      e0[r] = __expf(a0 - mnew);
      e1[r] = __expf(a1 - mnew);
      float rs = e0[r] + e1[r];
      rs += __shfl_xor(rs, 1, 32);
      rs += __shfl_xor(rs, 2, 32);
      rs += __shfl_xor(rs, 4, 32);
      rs += __shfl_xor(rs, 8, 32);
      lrun[r] = lrun[r] * alpha + rs;
      mrun[r] = mnew;
#pragma unroll
      for (int t = 0; t < 4; ++t) o[t][r] *= alpha;
    }

    // re-layout exp(S) (C-fragment) -> A-fragment through LDS
    __syncthreads();
#pragma unroll
    for (int r = 0; r < 8; ++r) {
      myLds[(r + hi * 8) * 33 + mrow]      = e0[r];
      myLds[(r + hi * 8) * 33 + 16 + mrow] = e1[r];
    }
    __syncthreads();
    Frag ap;
#pragma unroll
    for (int i = 0; i < 8; ++i) {
      const int kk = hi * 8 + 2 * i + ((i >= 4) ? 8 : 0);   // A elem pairs (2i,2i+1)
      const float f0 = myLds[mrow * 33 + kk];
      const float f1 = myLds[mrow * 33 + kk + 1];
      ap.u[i] = (u32)f2bf(f0) | ((u32)f2bf(f1) << 16);
    }

    // P (16x32 keys) @ V (32 keys x 64 dims): 4 WMMAs from the LDS V tile
#pragma unroll
    for (int t = 0; t < 4; ++t) {
      Frag bv;
      const u16* vp = vTl + (t * 16 + mrow) * 32 + hi * 8;
      load8(bv, 0, vp);
      load8(bv, 1, vp + 16);
      o[t] = wmma_bf16(ap, bv, o[t]);
    }
  }

  // normalize and store ctx[token, dmodel] bf16
  const int b = bh >> 4, h = bh & 15;
#pragma unroll
  for (int r = 0; r < 8; ++r) {
    const float inv = 1.0f / lrun[r];
    const size_t row = (size_t)(b * SEQ + qt * 16 + r + hi * 8);
#pragma unroll
    for (int t = 0; t < 4; ++t) {
      ctx[row * D_MODEL + h * D_K + t * 16 + mrow] = f2bf(o[t][r] * inv);
    }
  }
}

// ---------------------------------------------------------------------------
extern "C" void kernel_launch(void* const* d_in, const int* in_sizes, int n_in,
                              void* d_out, int out_size, void* d_ws, size_t ws_size,
                              hipStream_t stream) {
  (void)in_sizes; (void)n_in; (void)out_size; (void)ws_size;
  const float* Q  = (const float*)d_in[0];
  const float* K  = (const float*)d_in[1];
  const float* V  = (const float*)d_in[2];
  const float* Wq = (const float*)d_in[3];
  const float* bq = (const float*)d_in[4];
  const float* Wk = (const float*)d_in[5];
  const float* bk = (const float*)d_in[6];
  const float* Wv = (const float*)d_in[7];
  const float* bv = (const float*)d_in[8];
  const float* Wo = (const float*)d_in[9];
  const float* bo = (const float*)d_in[10];

  // workspace layout (64 MB total)
  char* ws = (char*)d_ws;
  const size_t ACT = (size_t)MTOK * D_MODEL * sizeof(u16);      // 8 MB
  const size_t WT  = (size_t)D_MODEL * D_MODEL * sizeof(u16);   // 2 MB
  u16* Qb  = (u16*)(ws);
  u16* Kb  = (u16*)(ws + ACT);
  u16* Vb  = (u16*)(ws + 2 * ACT);
  u16* WqT = (u16*)(ws + 3 * ACT);
  u16* WkT = (u16*)(ws + 3 * ACT + WT);
  u16* WvT = (u16*)(ws + 3 * ACT + 2 * WT);
  u16* WoT = (u16*)(ws + 3 * ACT + 3 * WT);
  u16* qpj = (u16*)(ws + 3 * ACT + 4 * WT);
  u16* kpj = (u16*)(ws + 4 * ACT + 4 * WT);
  u16* vTj = (u16*)(ws + 5 * ACT + 4 * WT);
  u16* ctx = (u16*)(ws + 6 * ACT + 4 * WT);

  const int nAct = MTOK * D_MODEL;            // 4 Mi elements
  const int nW   = D_MODEL * D_MODEL;         // 1 Mi elements

  // 1) convert activations + weights to bf16 (weights transposed to [N,K])
  mha_cvt_bf16<<<nAct / 256, 256, 0, stream>>>(Q, Qb, nAct);
  mha_cvt_bf16<<<nAct / 256, 256, 0, stream>>>(K, Kb, nAct);
  mha_cvt_bf16<<<nAct / 256, 256, 0, stream>>>(V, Vb, nAct);
  mha_cvtT_bf16<<<nW / 256, 256, 0, stream>>>(Wq, WqT, D_MODEL, D_MODEL);
  mha_cvtT_bf16<<<nW / 256, 256, 0, stream>>>(Wk, WkT, D_MODEL, D_MODEL);
  mha_cvtT_bf16<<<nW / 256, 256, 0, stream>>>(Wv, WvT, D_MODEL, D_MODEL);
  mha_cvtT_bf16<<<nW / 256, 256, 0, stream>>>(Wo, WoT, D_MODEL, D_MODEL);

  // 2) projections: waves = (M/16)*(N/64) = 4096 -> 1024 blocks of 4 waves
  dim3 gemmGrid(1024), gemmBlk(128);
  mha_gemm_bf16<0, u16><<<gemmGrid, gemmBlk, 0, stream>>>(Qb, WqT, bq, qpj, MTOK, D_MODEL, D_MODEL);
  mha_gemm_bf16<0, u16><<<gemmGrid, gemmBlk, 0, stream>>>(Kb, WkT, bk, kpj, MTOK, D_MODEL, D_MODEL);
  mha_gemm_bf16<1, u16><<<gemmGrid, gemmBlk, 0, stream>>>(Vb, WvT, bv, vTj, MTOK, D_MODEL, D_MODEL);

  // 3) flash attention: B*H*(S/16) = 4096 waves -> 2048 blocks of 2 waves
  mha_flash<<<dim3(2048), dim3(64), 0, stream>>>(qpj, kpj, vTj, ctx);

  // 4) output projection -> fp32 d_out
  mha_gemm_bf16<2, float><<<gemmGrid, gemmBlk, 0, stream>>>(ctx, WoT, bo, (float*)d_out, MTOK, D_MODEL, D_MODEL);
}
